// create_lstm_86019605004915
// MI455X (gfx1250) — matrix operand order repaired and checked by reading, not compiled, not run on hardware
//
#include <hip/hip_runtime.h>
#include <hip/hip_bf16.h>

// ---------------------------------------------------------------------------
// LSTM-style scan for MI455X (gfx1250, wave32, WMMA).
//   TS=256, BS=128, DIN=512, DOUT=512.
//   Fused per-step GEMM:  g[128,2048] = [x_t | H] (128x1024 bf16) x W^T,
//   W packed as [n=2048][k=1024] bf16 (k<512 from Wx, k>=512 from Wh).
//   WMMA: v_wmma_f32_16x16x32_bf16, f32 accumulate.
//   Per-block B tile (4 gates x 16 cols x K) staged in LDS in fragment
//   order (conflict-free ds_load_b128), A double-buffered in registers.
// ---------------------------------------------------------------------------

#define TS   256
#define BS   128
#define DIN  512
#define DOUT 512
#define NGATE 4
#define NTOT (NGATE * DOUT)   // 2048
#define KTOT (DIN + DOUT)     // 1024
#define KCHUNKS (KTOT / 32)   // 32
#define B_LDS_BYTES (NGATE * KCHUNKS * 32 * 32)   // 128 frags * 1KB = 131072

typedef __attribute__((ext_vector_type(16))) __bf16 v16bf;
typedef __attribute__((ext_vector_type(8)))  __bf16 v8bf;
typedef __attribute__((ext_vector_type(8)))  float  v8f;

// --------------------------- workspace layout (bytes) ----------------------
// W_bf16 : [2048][1024] bf16            4,194,304
// b_comb : [2048] f32                       8,192
// x_bf16 : [256][128][512] bf16        33,554,432
// H_bf16 : [128][512] bf16                131,072
// C_f32  : [128][512] f32                 262,144
#define OFF_W    0
#define OFF_B    (OFF_W + (size_t)NTOT * KTOT * 2)
#define OFF_X    (OFF_B + (size_t)NTOT * 4)
#define OFF_H    (OFF_X + (size_t)TS * BS * DIN * 2)
#define OFF_C    (OFF_H + (size_t)BS * DOUT * 2)

__device__ __forceinline__ float sigmoidf_(float x) {
    return 1.0f / (1.0f + __expf(-x));
}

// ------------------- pack Wx|Wh into bf16 W[n][k] row-major ----------------
__global__ void pack_W_kernel(const float* __restrict__ Wx,
                              const float* __restrict__ Wh,
                              __bf16* __restrict__ W) {
    int idx = blockIdx.x * blockDim.x + threadIdx.x;   // n*KTOT + k
    if (idx >= NTOT * KTOT) return;
    int n = idx / KTOT;          // n = g*512 + o
    int k = idx - n * KTOT;
    float v;
    if (k < DIN) v = Wx[(size_t)n * DIN + k];          // Wx[g][o][k]
    else         v = Wh[(size_t)n * DOUT + (k - DIN)]; // Wh[g][o][k-512]
    W[idx] = (__bf16)v;
}

__global__ void pack_b_kernel(const float* __restrict__ bx,
                              const float* __restrict__ bh,
                              float* __restrict__ b) {
    int n = blockIdx.x * blockDim.x + threadIdx.x;
    if (n >= NTOT) return;
    b[n] = bx[n] + bh[n];
}

// ------------------------------ x -> bf16 ----------------------------------
__global__ void cvt_x_kernel(const float* __restrict__ x,
                             __bf16* __restrict__ xb) {
    int idx = blockIdx.x * blockDim.x + threadIdx.x;
    if (idx >= TS * BS * DIN) return;
    xb[idx] = (__bf16)x[idx];
}

// ----------------------------- init H=0, C=0 -------------------------------
__global__ void init_state_kernel(__bf16* __restrict__ Hb,
                                  float* __restrict__ C) {
    int idx = blockIdx.x * blockDim.x + threadIdx.x;
    if (idx >= BS * DOUT) return;
    Hb[idx] = (__bf16)0.0f;
    C[idx]  = 0.0f;
}

// ---------------------------------------------------------------------------
// One timestep. grid = 32 blocks (16 output cols each), block = 256 thr
// (8 waves). Wave w owns rows 16w..16w+15; block b owns cols 16b..16b+15,
// and computes all 4 gate columns so the cell update fuses in-register.
//
// LDS: B tile in fragment order. Fragment f = kc*4 + g is 1KB: the 32B
// slice lane L consumes for v_wmma (B 32x16 bf16 layout: lanes 0-15 hold
// K=kb+{0..7,16..23} of col c0+L, lanes 16-31 hold K=kb+{8..15,24..31} of col c0+L-16).
// ---------------------------------------------------------------------------
__global__ __launch_bounds__(256)
void lstm_step_kernel(const __bf16* __restrict__ xb,   // x_bf16 base
                      const __bf16* __restrict__ W,    // [2048][1024]
                      const float*  __restrict__ bias, // [2048]
                      __bf16* __restrict__ Hb,         // [128][512] bf16
                      float*  __restrict__ C,          // [128][512] f32
                      float*  __restrict__ Hout,       // d_out
                      int t) {
    extern __shared__ __align__(32) char smem[];       // 131072 bytes

    const int lane = threadIdx.x & 31;
    const int wv   = threadIdx.x >> 5;      // 0..7 -> M tile (16 rows)
    const int l    = lane & 15;
    const int hi   = lane >> 4;             // 0: lanes 0-15, 1: lanes 16-31
    const int c0   = blockIdx.x * 16;       // output col base, 0..496

    const __bf16* xt = xb + (size_t)t * BS * DIN;

    // ---------------- stage B tile into LDS (once per step) ------------
    // 128 fragments; wave w stages fragments w, w+8, ..., w+120.
    // Per fragment each lane moves its own 32B WMMA slice.
#pragma unroll 4
    for (int j = 0; j < 16; ++j) {
        const int f  = wv + 8 * j;          // 0..127  (f = kc*4 + g)
        const int g  = f & 3;
        const int kc = f >> 2;
        const int n  = g * DOUT + c0 + l;
        const int kq = kc * 32 + hi * 8;
        const v8bf vlo = *(const v8bf*)(W + (size_t)n * KTOT + kq);
        const v8bf vhi = *(const v8bf*)(W + (size_t)n * KTOT + kq + 16);
        v16bf v = __builtin_shufflevector(vlo, vhi,
            0, 1, 2, 3, 4, 5, 6, 7, 8, 9, 10, 11, 12, 13, 14, 15);
        *(v16bf*)(smem + f * 1024 + lane * 32) = v;
    }
    __syncthreads();

    // ---------------- main GEMM loop -----------------------------------
    v8f acc0 = {}, acc1 = {}, acc2 = {}, acc3 = {};

    // A-fragment row for this lane (CDNA5 16-bit A 16x32 layout):
    //   lanes 0-15 : M=lane,    K = kb+{0..7}  and kb+{16..23}
    //   lanes 16-31: M=lane-16, K = kb+{8..15} and kb+{24..31}
    const int arow = wv * 16 + l;

    // preload A chunk 0 (kb = 0 < DIN -> from x_t)
    {
        const __bf16* ap0 = xt + (size_t)arow * 512 + hi * 8;
        __builtin_prefetch(ap0 + 32, 0, 0);
    }
    v8bf alo = *(const v8bf*)(xt + (size_t)arow * 512 + hi * 8);
    v8bf ahv = *(const v8bf*)(xt + (size_t)arow * 512 + hi * 8 + 16);

    for (int kc = 0; kc < KCHUNKS; ++kc) {
        const v16bf afrag = __builtin_shufflevector(alo, ahv,
            0, 1, 2, 3, 4, 5, 6, 7, 8, 9, 10, 11, 12, 13, 14, 15);

        // double-buffer: issue next chunk's A loads before this chunk's WMMAs
        if (kc + 1 < KCHUNKS) {
            const int kb = (kc + 1) * 32;
            const __bf16* Abase = (kb < DIN) ? xt : Hb;
            const int krow      = (kb < DIN) ? kb : (kb - DIN);
            const __bf16* ap = Abase + (size_t)arow * 512 + krow + hi * 8;
            __builtin_prefetch(ap + 32, 0, 0);   // global_prefetch_b8
            alo = *(const v8bf*)(ap);
            ahv = *(const v8bf*)(ap + 16);
        }

        // B fragments from LDS: base + immediate offsets, conflict-free.
        const char* bb = smem + kc * 4096 + lane * 32;
        const v16bf b0 = *(const v16bf*)(bb);
        const v16bf b1 = *(const v16bf*)(bb + 1024);
        const v16bf b2 = *(const v16bf*)(bb + 2048);
        const v16bf b3 = *(const v16bf*)(bb + 3072);

        acc0 = __builtin_amdgcn_wmma_f32_16x16x32_bf16(
            false, afrag, false, b0, (short)0, acc0, false, false);
        acc1 = __builtin_amdgcn_wmma_f32_16x16x32_bf16(
            false, afrag, false, b1, (short)0, acc1, false, false);
        acc2 = __builtin_amdgcn_wmma_f32_16x16x32_bf16(
            false, afrag, false, b2, (short)0, acc2, false, false);
        acc3 = __builtin_amdgcn_wmma_f32_16x16x32_bf16(
            false, afrag, false, b3, (short)0, acc3, false, false);
    }

    // ---------------- fused gate epilogue -------------------------------
    // C/D layout: VGPR r holds (M = 16*wv + r + 8*hi, N = c0 + (lane&15)).
    const int col = c0 + l;
    const float b0 = bias[0 * DOUT + col];
    const float b1 = bias[1 * DOUT + col];
    const float b2 = bias[2 * DOUT + col];
    const float b3 = bias[3 * DOUT + col];
    const int mbase = wv * 16 + hi * 8;
    float* outT = Hout + (size_t)t * BS * DOUT;

#pragma unroll
    for (int r = 0; r < 8; ++r) {
        const int m = mbase + r;
        const size_t idx = (size_t)m * DOUT + col;
        const float f  = sigmoidf_(acc0[r] + b0);
        const float i  = tanhf(acc1[r] + b1);
        const float cg = sigmoidf_(acc2[r] + b2);
        const float og = tanhf(acc3[r] + b3);
        const float cn = i * cg + f * C[idx];
        const float h  = og * tanhf(cn);
        C[idx]    = cn;
        Hb[idx]   = (__bf16)h;   // bf16 feedback for next step's GEMM
        outT[idx] = h;           // f32 result
    }
}

// ---------------------------------------------------------------------------
extern "C" void kernel_launch(void* const* d_in, const int* in_sizes, int n_in,
                              void* d_out, int out_size, void* d_ws, size_t ws_size,
                              hipStream_t stream) {
    const float* x  = (const float*)d_in[0];  // [256,128,512]
    const float* Wx = (const float*)d_in[1];  // [4,512,512]
    const float* bx = (const float*)d_in[2];  // [4,512]
    const float* Wh = (const float*)d_in[3];  // [4,512,512]
    const float* bh = (const float*)d_in[4];  // [4,512]
    float* out = (float*)d_out;               // [256,128,512]

    char* ws = (char*)d_ws;
    __bf16* W   = (__bf16*)(ws + OFF_W);
    float*  bcb = (float*) (ws + OFF_B);
    __bf16* xb  = (__bf16*)(ws + OFF_X);
    __bf16* Hb  = (__bf16*)(ws + OFF_H);
    float*  C   = (float*) (ws + OFF_C);

    {
        int n = NTOT * KTOT;
        pack_W_kernel<<<(n + 255) / 256, 256, 0, stream>>>(Wx, Wh, W);
    }
    pack_b_kernel<<<(NTOT + 255) / 256, 256, 0, stream>>>(bx, bh, bcb);
    {
        int n = TS * BS * DIN;
        cvt_x_kernel<<<(n + 255) / 256, 256, 0, stream>>>(x, xb);
    }
    {
        int n = BS * DOUT;
        init_state_kernel<<<(n + 255) / 256, 256, 0, stream>>>(Hb, C);
    }

    // Sequential scan; stream ordering provides the cross-step dependency.
    for (int t = 0; t < TS; ++t) {
        lstm_step_kernel<<<dim3(DOUT / 16), dim3(256), B_LDS_BYTES, stream>>>(
            xb, W, bcb, Hb, C, out, t);
    }
}
